// ModernBertEncoderLayer_55319178772493
// MI455X (gfx1250) — compile-verified, matrix operand order checked by reading, not run on hardware
//
#include <hip/hip_runtime.h>
#include <hip/hip_bf16.h>

typedef __bf16 bf16;
typedef __bf16 bf16x8  __attribute__((ext_vector_type(8)));
typedef __bf16 bf16x16 __attribute__((ext_vector_type(16)));
typedef float  f32x8   __attribute__((ext_vector_type(8)));

#define WMMA_BF16(A_, B_, C_) \
  __builtin_amdgcn_wmma_f32_16x16x32_bf16(false, (A_), false, (B_), (short)0, (C_), false, false)

// ---- async global->LDS copy (CDNA5), with safe fallback --------------------
#if defined(__has_builtin)
#if __has_builtin(__builtin_amdgcn_global_load_async_to_lds_b128)
#define HAVE_ASYNC 1
#endif
#endif
#ifndef HAVE_ASYNC
#define HAVE_ASYNC 0
#endif

#if HAVE_ASYNC
// builtin signature (from toolchain diagnostic):
//   void (v4i addrspace(1)*, v4i addrspace(3)*, imm int offset, imm int cpol)
// where v4i is the GCC-vector 'int __attribute__((vector_size(16)))'.
typedef int v4i_gcc __attribute__((vector_size(16)));
typedef __attribute__((address_space(1))) v4i_gcc glb_v4i;
typedef __attribute__((address_space(3))) v4i_gcc lds_v4i;
__device__ __forceinline__ void cp128(bf16* dst_lds, const bf16* src_glb) {
  __builtin_amdgcn_global_load_async_to_lds_b128(
      (glb_v4i*)src_glb, (lds_v4i*)dst_lds, 0, 0);
}
#if __has_builtin(__builtin_amdgcn_s_wait_asynccnt)
#define ASYNC_WAIT(n) __builtin_amdgcn_s_wait_asynccnt(n)
#else
#define ASYNC_WAIT(n) asm volatile("s_wait_asynccnt %0" ::"i"(n))
#endif
#else
__device__ __forceinline__ void cp128(bf16* dst_lds, const bf16* src_glb) {
  *(float4*)dst_lds = *(const float4*)src_glb;
}
#define ASYNC_WAIT(n)
#endif

__device__ __forceinline__ f32x8 zero8() {
  f32x8 z;
#pragma unroll
  for (int i = 0; i < 8; ++i) z[i] = 0.0f;
  return z;
}

// Load a 16x32 bf16 fragment (A-layout; B mirrored with lane=N).
// lane<16: M=lane, K = {0..7, 16..23}; lane>=16: M=lane-16, K = {8..15, 24..31}
__device__ __forceinline__ bf16x16 load_frag_row(const bf16* base, int ld) {
  const int lane = threadIdx.x & 31;
  const int m    = lane & 15;
  const int half = lane >> 4;
  const bf16* p = base + m * ld + half * 8;
  bf16x8 lo = *(const bf16x8*)(p);
  bf16x8 hi = *(const bf16x8*)(p + 16);
  bf16x16 r;
#pragma unroll
  for (int i = 0; i < 8; ++i) { r[i] = lo[i]; r[i + 8] = hi[i]; }
  return r;
}

__device__ __forceinline__ float block_reduce_sum(float v, float* red) {
  int tid = threadIdx.x;
  red[tid] = v;
  __syncthreads();
#pragma unroll
  for (int s = 128; s > 0; s >>= 1) {
    if (tid < s) red[tid] += red[tid + s];
    __syncthreads();
  }
  float r = red[0];
  __syncthreads();
  return r;
}

// ---------------------------------------------------------------------------
// Weight transpose + f32->bf16 convert:  Wt[n][k] = (bf16) W[k][n]
// ---------------------------------------------------------------------------
__global__ void transpose_cvt(const float* __restrict__ W, bf16* __restrict__ Wt,
                              int K, int Nn) {
  __shared__ float t[32][33];
  int k0 = blockIdx.x * 32, n0 = blockIdx.y * 32;
  int tx = threadIdx.x, ty = threadIdx.y;
#pragma unroll
  for (int i = 0; i < 4; ++i)
    t[ty + i * 8][tx] = W[(size_t)(k0 + ty + i * 8) * Nn + n0 + tx];
  __syncthreads();
#pragma unroll
  for (int i = 0; i < 4; ++i)
    Wt[(size_t)(n0 + ty + i * 8) * K + k0 + tx] = (bf16)t[tx][ty + i * 8];
}

// ---------------------------------------------------------------------------
// mhc coefficients + x_in + LayerNorm.  One block per (b,s), 256 threads.
// ---------------------------------------------------------------------------
__global__ __launch_bounds__(256) void coeff_kernel(
    const float* __restrict__ x, const float* __restrict__ phi,
    const float* __restrict__ bvec, const float* __restrict__ alphas,
    const float* __restrict__ gamma, bf16* __restrict__ hln,
    float* __restrict__ hpost, float* __restrict__ hres) {
  __shared__ float xf[3072];
  __shared__ float xin[768];
  __shared__ float red[256];
  __shared__ float mixsh[24];
  __shared__ float hpre[4];
  const int tid = threadIdx.x, lane = tid & 31, w = tid >> 5;
  const int bs = blockIdx.x;
  const float* xr = x + (size_t)bs * 3072;

  float ss = 0.0f;
#pragma unroll
  for (int it = 0; it < 12; ++it) {
    int i = tid + it * 256;
    float v = xr[i];
    xf[i] = v;
    ss += v * v;
  }
  __syncthreads();
  float tot = block_reduce_sum(ss, red);
  float invr = rsqrtf(tot * (1.0f / 3072.0f) + 1e-6f);

  // mix = xf @ phi : wave w computes cols 3w..3w+2
  float p0 = 0.f, p1 = 0.f, p2 = 0.f;
  const float* ph = phi + w * 3;
  for (int j = lane; j < 3072; j += 32) {
    float v = xf[j];
    p0 += v * ph[j * 24 + 0];
    p1 += v * ph[j * 24 + 1];
    p2 += v * ph[j * 24 + 2];
  }
#pragma unroll
  for (int d = 16; d >= 1; d >>= 1) {
    p0 += __shfl_xor(p0, d, 32);
    p1 += __shfl_xor(p1, d, 32);
    p2 += __shfl_xor(p2, d, 32);
  }
  if (lane == 0) { mixsh[w*3] = p0; mixsh[w*3+1] = p1; mixsh[w*3+2] = p2; }
  __syncthreads();

  if (tid < 24) {
    float mv = mixsh[tid] * invr;
    if (tid < 4)
      hpre[tid] = 1.0f / (1.0f + __expf(-(mv * alphas[0] + bvec[tid])));
    else if (tid < 8)
      hpost[(size_t)bs * 4 + tid - 4] =
          2.0f / (1.0f + __expf(-(mv * alphas[1] + bvec[tid])));
    else
      mixsh[tid] = mv * alphas[2] + bvec[tid];
  }
  __syncthreads();

  if (tid == 0) {  // sinkhorn on 4x4
    float r[16];
#pragma unroll
    for (int i = 0; i < 4; ++i) {
      float mx = -1e30f;
      for (int j = 0; j < 4; ++j) mx = fmaxf(mx, mixsh[8 + i * 4 + j]);
      for (int j = 0; j < 4; ++j) r[i * 4 + j] = __expf(mixsh[8 + i * 4 + j] - mx);
    }
    for (int t = 0; t < 20; ++t) {
      for (int i = 0; i < 4; ++i) {
        float s = r[i*4] + r[i*4+1] + r[i*4+2] + r[i*4+3] + 1e-6f;
        float inv = 1.0f / s;
        for (int j = 0; j < 4; ++j) r[i * 4 + j] *= inv;
      }
      for (int j = 0; j < 4; ++j) {
        float s = r[j] + r[4 + j] + r[8 + j] + r[12 + j] + 1e-6f;
        float inv = 1.0f / s;
        for (int i = 0; i < 4; ++i) r[i * 4 + j] *= inv;
      }
    }
    for (int i = 0; i < 16; ++i) hres[(size_t)bs * 16 + i] = r[i];
  }

  // x_in = h_pre . x   then LayerNorm * gamma -> bf16
  float msum = 0.0f;
#pragma unroll
  for (int it = 0; it < 3; ++it) {
    int c = tid + it * 256;
    float v = hpre[0]*xf[c] + hpre[1]*xf[768+c] + hpre[2]*xf[1536+c] + hpre[3]*xf[2304+c];
    xin[c] = v;
    msum += v;
  }
  float mu = block_reduce_sum(msum, red) * (1.0f / 768.0f);
  float vs = 0.0f;
#pragma unroll
  for (int it = 0; it < 3; ++it) {
    int c = tid + it * 256;
    float d = xin[c] - mu;
    vs += d * d;
  }
  float var = block_reduce_sum(vs, red) * (1.0f / 768.0f);
  float rstd = rsqrtf(var + 1e-5f);
#pragma unroll
  for (int it = 0; it < 3; ++it) {
    int c = tid + it * 256;
    hln[(size_t)bs * 768 + c] = (bf16)((xin[c] - mu) * rstd * gamma[c]);
  }
}

// ---------------------------------------------------------------------------
// bf16 WMMA GEMM:  C[M][N] (f32) = A[M][K] (bf16) * Bt[N][K]^T (bf16)
// BM=128, BN=64, BK=32; 8 waves; each wave 32x32 (2x2 WMMA tiles)
// Double-buffered: async global->LDS (ASYNCcnt) or reg-staged fallback.
// ---------------------------------------------------------------------------
__global__ __launch_bounds__(256) void gemm_bf16(
    const bf16* __restrict__ A, const bf16* __restrict__ Bt,
    float* __restrict__ C, int M, int Nn, int K) {
  __shared__ alignas(16) bf16 As[2][128 * 40];
  __shared__ alignas(16) bf16 Bs[2][64 * 40];
  const int tid = threadIdx.x, lane = tid & 31, wave = tid >> 5;
  const int wm = wave >> 1, wn = wave & 1;
  const int row0 = blockIdx.y * 128, col0 = blockIdx.x * 64;
  const int ra0 = tid >> 2, ca = (tid & 3) * 8;   // A chunks: rows ra0 and ra0+64
  const int ntiles = K >> 5;

  const bf16* Ab = A + (size_t)row0 * K;
  const bf16* Bb = Bt + (size_t)col0 * K;

  f32x8 acc[2][2];
  acc[0][0] = zero8(); acc[0][1] = zero8(); acc[1][0] = zero8(); acc[1][1] = zero8();

#if HAVE_ASYNC
  // prologue: tile 0 -> buffer 0
  cp128(&As[0][ra0 * 40 + ca],        Ab + (size_t)ra0 * K + ca);
  cp128(&As[0][(ra0 + 64) * 40 + ca], Ab + (size_t)(ra0 + 64) * K + ca);
  cp128(&Bs[0][ra0 * 40 + ca],        Bb + (size_t)ra0 * K + ca);
  for (int i = 0; i < ntiles; ++i) {
    const int cur = i & 1;
    if (i + 1 < ntiles) {
      const int k0 = (i + 1) << 5;
      cp128(&As[cur ^ 1][ra0 * 40 + ca],        Ab + (size_t)ra0 * K + k0 + ca);
      cp128(&As[cur ^ 1][(ra0 + 64) * 40 + ca], Ab + (size_t)(ra0 + 64) * K + k0 + ca);
      cp128(&Bs[cur ^ 1][ra0 * 40 + ca],        Bb + (size_t)ra0 * K + k0 + ca);
      ASYNC_WAIT(3);   // tile i done; tile i+1 may stay in flight
    } else {
      ASYNC_WAIT(0);
    }
    __syncthreads();
    bf16x16 af[2], bfv[2];
    af[0]  = load_frag_row(&As[cur][(wm * 32 + 0)  * 40], 40);
    af[1]  = load_frag_row(&As[cur][(wm * 32 + 16) * 40], 40);
    bfv[0] = load_frag_row(&Bs[cur][(wn * 32 + 0)  * 40], 40);
    bfv[1] = load_frag_row(&Bs[cur][(wn * 32 + 16) * 40], 40);
#pragma unroll
    for (int tm = 0; tm < 2; ++tm)
#pragma unroll
      for (int tn = 0; tn < 2; ++tn)
        acc[tm][tn] = WMMA_BF16(af[tm], bfv[tn], acc[tm][tn]);
    __syncthreads();
  }
#else
  // register-staged double buffer: next tile's global loads overlap compute
  float4 pa0 = *(const float4*)(Ab + (size_t)ra0 * K + ca);
  float4 pa1 = *(const float4*)(Ab + (size_t)(ra0 + 64) * K + ca);
  float4 pb  = *(const float4*)(Bb + (size_t)ra0 * K + ca);
  for (int i = 0; i < ntiles; ++i) {
    *(float4*)&As[0][ra0 * 40 + ca]        = pa0;
    *(float4*)&As[0][(ra0 + 64) * 40 + ca] = pa1;
    *(float4*)&Bs[0][ra0 * 40 + ca]        = pb;
    __syncthreads();
    if (i + 1 < ntiles) {
      const int k0 = (i + 1) << 5;
      pa0 = *(const float4*)(Ab + (size_t)ra0 * K + k0 + ca);
      pa1 = *(const float4*)(Ab + (size_t)(ra0 + 64) * K + k0 + ca);
      pb  = *(const float4*)(Bb + (size_t)ra0 * K + k0 + ca);
    }
    bf16x16 af[2], bfv[2];
    af[0]  = load_frag_row(&As[0][(wm * 32 + 0)  * 40], 40);
    af[1]  = load_frag_row(&As[0][(wm * 32 + 16) * 40], 40);
    bfv[0] = load_frag_row(&Bs[0][(wn * 32 + 0)  * 40], 40);
    bfv[1] = load_frag_row(&Bs[0][(wn * 32 + 16) * 40], 40);
#pragma unroll
    for (int tm = 0; tm < 2; ++tm)
#pragma unroll
      for (int tn = 0; tn < 2; ++tn)
        acc[tm][tn] = WMMA_BF16(af[tm], bfv[tn], acc[tm][tn]);
    __syncthreads();
  }
#endif

  const int col = lane & 15, half = lane >> 4;
  float* Cp = C + (size_t)(row0 + wm * 32 + half * 8) * Nn + col0 + wn * 32 + col;
#pragma unroll
  for (int tm = 0; tm < 2; ++tm)
#pragma unroll
    for (int vv = 0; vv < 8; ++vv) {
      float* r = Cp + (size_t)(tm * 16 + vv) * Nn;
      r[0]  = acc[tm][0][vv];
      r[16] = acc[tm][1][vv];
    }
}

// ---------------------------------------------------------------------------
// RoPE + scatter qkv (f32 [4096][2304]) -> q,k bf16 [B,H,S,D]; v^T bf16 [B,H,D,S]
// ---------------------------------------------------------------------------
__global__ void rope_kernel(const float* __restrict__ qkv,
                            const float* __restrict__ cosb,
                            const float* __restrict__ sinb,
                            bf16* __restrict__ qb, bf16* __restrict__ kb,
                            bf16* __restrict__ vtb) {
  const int bs = blockIdx.x;
  const int b = bs >> 11, s = bs & 2047;
  const float* row = qkv + (size_t)bs * 2304;
#pragma unroll
  for (int it = 0; it < 9; ++it) {
    int i = threadIdx.x + it * 256;
    int e = i / 768, r = i - e * 768, h = r >> 6, d = r & 63;
    float val = row[i];
    if (e == 2) {
      vtb[((size_t)(b * 12 + h) * 64 + d) * 2048 + s] = (bf16)val;
    } else {
      float other = row[e * 768 + h * 64 + ((d < 32) ? d + 32 : d - 32)];
      float rh = (d < 32) ? -other : other;
      float o = val * cosb[s * 64 + d] + rh * sinb[s * 64 + d];
      bf16* dst = (e == 0) ? qb : kb;
      dst[((size_t)(b * 12 + h) * 2048 + s) * 64 + d] = (bf16)o;
    }
  }
}

// ---------------------------------------------------------------------------
// Flash attention per (b,h).  grid: (S/64, B*H), block 128 (4 waves x 16 rows)
// K/V tiles double-buffered in LDS (async path) or reg-staged (fallback).
// ---------------------------------------------------------------------------
__global__ __launch_bounds__(128) void attn_kernel(
    const bf16* __restrict__ qb, const bf16* __restrict__ kb,
    const bf16* __restrict__ vtb, bf16* __restrict__ obf) {
  __shared__ alignas(16) bf16 Ks[2][32 * 72];
  __shared__ alignas(16) bf16 Vs[2][64 * 40];
  __shared__ alignas(16) bf16 Ps[4 * 16 * 32];
  const int tid = threadIdx.x, lane = tid & 31, w = tid >> 5;
  const int bh = blockIdx.y, b = bh / 12, h = bh % 12;
  const bf16* q   = qb  + (size_t)bh * 2048 * 64;
  const bf16* kg  = kb  + (size_t)bh * 2048 * 64;
  const bf16* vg  = vtb + (size_t)bh * 64 * 2048;
  const int q0 = blockIdx.x * 64 + w * 16;

  // K-tile chunks: 32x64 = 256 chunks of 8; thread does rk0 and rk0+16 rows
  const int rk0 = tid >> 3, ck = (tid & 7) * 8;
  // V-tile chunks: 64x32 = 256 chunks of 8; thread does rv0 and rv0+32 rows
  const int rv0 = tid >> 2, cv = (tid & 3) * 8;

  bf16x16 qf[2];
  qf[0] = load_frag_row(q + (size_t)q0 * 64 + 0, 64);
  qf[1] = load_frag_row(q + (size_t)q0 * 64 + 32, 64);

  f32x8 accO[4];
  float m_i[8], l_i[8];
#pragma unroll
  for (int i = 0; i < 4; ++i) accO[i] = zero8();
#pragma unroll
  for (int i = 0; i < 8; ++i) { m_i[i] = -1e30f; l_i[i] = 0.0f; }
  const float scale = 0.125f;  // 1/sqrt(64)

#if HAVE_ASYNC
  cp128(&Ks[0][rk0 * 72 + ck],        kg + (size_t)rk0 * 64 + ck);
  cp128(&Ks[0][(rk0 + 16) * 72 + ck], kg + (size_t)(rk0 + 16) * 64 + ck);
  cp128(&Vs[0][rv0 * 40 + cv],        vg + (size_t)rv0 * 2048 + cv);
  cp128(&Vs[0][(rv0 + 32) * 40 + cv], vg + (size_t)(rv0 + 32) * 2048 + cv);
#else
  float4 sk0 = *(const float4*)(kg + (size_t)rk0 * 64 + ck);
  float4 sk1 = *(const float4*)(kg + (size_t)(rk0 + 16) * 64 + ck);
  float4 sv0 = *(const float4*)(vg + (size_t)rv0 * 2048 + cv);
  float4 sv1 = *(const float4*)(vg + (size_t)(rv0 + 32) * 2048 + cv);
#endif

  for (int i = 0; i < 64; ++i) {      // 64 key tiles of 32
    const int cur = i & 1;
#if HAVE_ASYNC
    if (i + 1 < 64) {
      const int t1 = (i + 1) << 5;
      cp128(&Ks[cur ^ 1][rk0 * 72 + ck],        kg + (size_t)(t1 + rk0) * 64 + ck);
      cp128(&Ks[cur ^ 1][(rk0 + 16) * 72 + ck], kg + (size_t)(t1 + rk0 + 16) * 64 + ck);
      cp128(&Vs[cur ^ 1][rv0 * 40 + cv],        vg + (size_t)rv0 * 2048 + t1 + cv);
      cp128(&Vs[cur ^ 1][(rv0 + 32) * 40 + cv], vg + (size_t)(rv0 + 32) * 2048 + t1 + cv);
      ASYNC_WAIT(4);
    } else {
      ASYNC_WAIT(0);
    }
    __syncthreads();
    const bf16* Kt = &Ks[cur][0];
    const bf16* Vt = &Vs[cur][0];
#else
    *(float4*)&Ks[0][rk0 * 72 + ck]        = sk0;
    *(float4*)&Ks[0][(rk0 + 16) * 72 + ck] = sk1;
    *(float4*)&Vs[0][rv0 * 40 + cv]        = sv0;
    *(float4*)&Vs[0][(rv0 + 32) * 40 + cv] = sv1;
    __syncthreads();
    if (i + 1 < 64) {
      const int t1 = (i + 1) << 5;
      sk0 = *(const float4*)(kg + (size_t)(t1 + rk0) * 64 + ck);
      sk1 = *(const float4*)(kg + (size_t)(t1 + rk0 + 16) * 64 + ck);
      sv0 = *(const float4*)(vg + (size_t)rv0 * 2048 + t1 + cv);
      sv1 = *(const float4*)(vg + (size_t)(rv0 + 32) * 2048 + t1 + cv);
    }
    const bf16* Kt = &Ks[0][0];
    const bf16* Vt = &Vs[0][0];
#endif

    f32x8 accS[2];
    accS[0] = zero8(); accS[1] = zero8();
#pragma unroll
    for (int tt = 0; tt < 2; ++tt)
#pragma unroll
      for (int ks = 0; ks < 2; ++ks) {
        bf16x16 bk = load_frag_row(Kt + (tt * 16) * 72 + ks * 32, 72);
        accS[tt] = WMMA_BF16(qf[ks], bk, accS[tt]);
      }

    float fac[8];
#pragma unroll
    for (int vv = 0; vv < 8; ++vv) {
      float s0 = accS[0][vv] * scale, s1 = accS[1][vv] * scale;
      float mx = fmaxf(s0, s1);
#pragma unroll
      for (int d = 8; d >= 1; d >>= 1) mx = fmaxf(mx, __shfl_xor(mx, d, 32));
      float mn = fmaxf(m_i[vv], mx);
      fac[vv] = __expf(m_i[vv] - mn);
      float p0 = __expf(s0 - mn);
      float p1 = __expf(s1 - mn);
      accS[0][vv] = p0; accS[1][vv] = p1;
      float rs = p0 + p1;
#pragma unroll
      for (int d = 8; d >= 1; d >>= 1) rs += __shfl_xor(rs, d, 32);
      l_i[vv] = l_i[vv] * fac[vv] + rs;
      m_i[vv] = mn;
    }
#pragma unroll
    for (int dt = 0; dt < 4; ++dt)
#pragma unroll
      for (int vv = 0; vv < 8; ++vv) accO[dt][vv] *= fac[vv];

    // D-frag -> A-frag conversion via per-wave LDS round trip
    bf16* P = Ps + w * 512;
    const int col = lane & 15, half = lane >> 4;
#pragma unroll
    for (int vv = 0; vv < 8; ++vv) {
      int row = half * 8 + vv;
      P[row * 32 + col]      = (bf16)accS[0][vv];
      P[row * 32 + 16 + col] = (bf16)accS[1][vv];
    }
    bf16x16 pf = load_frag_row(P, 32);
#pragma unroll
    for (int dt = 0; dt < 4; ++dt) {
      bf16x16 vf = load_frag_row(Vt + dt * 16 * 40, 40);
      accO[dt] = WMMA_BF16(pf, vf, accO[dt]);
    }
    __syncthreads();
  }

  const int col = lane & 15, half = lane >> 4;
#pragma unroll
  for (int vv = 0; vv < 8; ++vv) {
    int s = blockIdx.x * 64 + w * 16 + half * 8 + vv;
    float inv = 1.0f / l_i[vv];
#pragma unroll
    for (int dt = 0; dt < 4; ++dt)
      obf[(size_t)(b * 2048 + s) * 768 + h * 64 + dt * 16 + col] =
          (bf16)(accO[dt][vv] * inv);
  }
}

// ---------------------------------------------------------------------------
// out[bs][i][c] = sum_j hres[i][j] * xsrc[bs][j][c] + hpost[i] * fout[bs][c]
// Safe when xsrc == out (per-thread column read-then-write).
// ---------------------------------------------------------------------------
__global__ void combine_kernel(const float* __restrict__ xsrc,
                               const float* __restrict__ fout,
                               const float* __restrict__ hres,
                               const float* __restrict__ hpost,
                               float* __restrict__ out) {
  const int bs = blockIdx.x;
  const float* hr = hres + (size_t)bs * 16;
  const float* hp = hpost + (size_t)bs * 4;
#pragma unroll
  for (int it = 0; it < 3; ++it) {
    int c = threadIdx.x + it * 256;
    float f = fout[(size_t)bs * 768 + c];
    float xj[4];
#pragma unroll
    for (int j = 0; j < 4; ++j) xj[j] = xsrc[(size_t)bs * 3072 + j * 768 + c];
#pragma unroll
    for (int i = 0; i < 4; ++i)
      out[(size_t)bs * 3072 + i * 768 + c] =
          hr[i*4+0]*xj[0] + hr[i*4+1]*xj[1] + hr[i*4+2]*xj[2] + hr[i*4+3]*xj[3] +
          hp[i] * f;
  }
}

// ---------------------------------------------------------------------------
// g[row][c] = silu(u[row][c]) * u[row][1152+c]  (bf16 out)
// ---------------------------------------------------------------------------
__global__ void silu_gate(const float* __restrict__ u, bf16* __restrict__ g) {
  size_t i = (size_t)blockIdx.x * 256 + threadIdx.x;
  size_t row = i / 1152, c = i % 1152;
  float a = u[row * 2304 + c];
  float bb = u[row * 2304 + 1152 + c];
  g[i] = (bf16)(a / (1.0f + __expf(-a)) * bb);
}

// ---------------------------------------------------------------------------
// Workspace layout (bytes)
// ---------------------------------------------------------------------------
#define OFF_W1T    0u          // Wqkv^T  bf16 [2304][768]
#define OFF_W2T    3538944u    // Wo^T    bf16 [768][768]
#define OFF_W3T    4718592u    // Wi^T    bf16 [2304][768]
#define OFF_W4T    8257536u    // Wmlp^T  bf16 [768][1152]
#define OFF_HLN    10027008u   // bf16 [4096][768] (shared by both stages)
#define OFF_HPOST1 16318464u
#define OFF_HRES1  16384000u
#define OFF_HPOST2 16646144u
#define OFF_HRES2  16711680u
#define OFF_BIG    16973824u   // f32 [4096][2304] (qkv, then u)
#define OFF_QB     54722560u   // bf16 [B,H,S,D] (later reused for gate g)
#define OFF_KB     61014016u
#define OFF_VT     67305472u
#define OFF_OBF    73596928u   // bf16 [4096][768]
#define OFF_FOUT   79888384u   // f32 [4096][768] (shared by both stages)

extern "C" void kernel_launch(void* const* d_in, const int* in_sizes, int n_in,
                              void* d_out, int out_size, void* d_ws, size_t ws_size,
                              hipStream_t stream) {
  const float* x          = (const float*)d_in[0];
  const float* cosb       = (const float*)d_in[1];
  const float* sinb       = (const float*)d_in[2];
  const float* gamma_attn = (const float*)d_in[3];
  const float* Wqkv       = (const float*)d_in[4];
  const float* Wo         = (const float*)d_in[5];
  const float* phi_a      = (const float*)d_in[6];
  const float* b_a        = (const float*)d_in[7];
  const float* alphas_a   = (const float*)d_in[8];
  const float* gamma_mlp  = (const float*)d_in[9];
  const float* Wi         = (const float*)d_in[10];
  const float* Wo_mlp     = (const float*)d_in[11];
  const float* phi_m      = (const float*)d_in[12];
  const float* b_m        = (const float*)d_in[13];
  const float* alphas_m   = (const float*)d_in[14];
  float* out = (float*)d_out;

  char* ws = (char*)d_ws;
  bf16*  W1t    = (bf16*)(ws + OFF_W1T);
  bf16*  W2t    = (bf16*)(ws + OFF_W2T);
  bf16*  W3t    = (bf16*)(ws + OFF_W3T);
  bf16*  W4t    = (bf16*)(ws + OFF_W4T);
  bf16*  hln    = (bf16*)(ws + OFF_HLN);
  float* hpost1 = (float*)(ws + OFF_HPOST1);
  float* hres1  = (float*)(ws + OFF_HRES1);
  float* hpost2 = (float*)(ws + OFF_HPOST2);
  float* hres2  = (float*)(ws + OFF_HRES2);
  float* big    = (float*)(ws + OFF_BIG);   // qkv f32, later u f32
  bf16*  qb     = (bf16*)(ws + OFF_QB);
  bf16*  kb     = (bf16*)(ws + OFF_KB);
  bf16*  vt     = (bf16*)(ws + OFF_VT);
  bf16*  obf    = (bf16*)(ws + OFF_OBF);
  float* fout   = (float*)(ws + OFF_FOUT);
  bf16*  gbuf   = (bf16*)(ws + OFF_QB);     // reuse q region for MLP gate

  dim3 tb(32, 8);
  transpose_cvt<<<dim3(24, 72), tb, 0, stream>>>(Wqkv,   W1t, 768,  2304);
  transpose_cvt<<<dim3(24, 24), tb, 0, stream>>>(Wo,     W2t, 768,  768);
  transpose_cvt<<<dim3(24, 72), tb, 0, stream>>>(Wi,     W3t, 768,  2304);
  transpose_cvt<<<dim3(36, 24), tb, 0, stream>>>(Wo_mlp, W4t, 1152, 768);

  // ---- stage 1: attention mhc ----
  coeff_kernel<<<4096, 256, 0, stream>>>(x, phi_a, b_a, alphas_a, gamma_attn,
                                         hln, hpost1, hres1);
  gemm_bf16<<<dim3(36, 32), 256, 0, stream>>>(hln, W1t, big, 4096, 2304, 768);
  rope_kernel<<<4096, 256, 0, stream>>>(big, cosb, sinb, qb, kb, vt);
  attn_kernel<<<dim3(32, 24), 128, 0, stream>>>(qb, kb, vt, obf);
  gemm_bf16<<<dim3(12, 32), 256, 0, stream>>>(obf, W2t, fout, 4096, 768, 768);
  combine_kernel<<<4096, 256, 0, stream>>>(x, fout, hres1, hpost1, out);

  // ---- stage 2: MLP mhc ----
  coeff_kernel<<<4096, 256, 0, stream>>>(out, phi_m, b_m, alphas_m, gamma_mlp,
                                         hln, hpost2, hres2);
  gemm_bf16<<<dim3(36, 32), 256, 0, stream>>>(hln, W3t, big, 4096, 2304, 768);
  silu_gate<<<18432, 256, 0, stream>>>(big, gbuf);
  gemm_bf16<<<dim3(12, 32), 256, 0, stream>>>(gbuf, W4t, fout, 4096, 768, 1152);
  combine_kernel<<<4096, 256, 0, stream>>>(out, fout, hres2, hpost2, out);
}
